// Tharvexal4MoE_83846351553131
// MI455X (gfx1250) — matrix operand order repaired and checked by reading, not compiled, not run on hardware
//
#include <hip/hip_runtime.h>
#include <stddef.h>
#include <stdint.h>

// ---------------------------------------------------------------------------
// Tharvexal4MoE for MI455X (gfx1250), wave32 + WMMA bf16.
// 32x32 wave tiles, 64x128 block tiles, register double-buffered K-loops.
// ---------------------------------------------------------------------------

typedef __bf16 bf16_t;
typedef __attribute__((ext_vector_type(16))) __bf16 bf16x16;
typedef __attribute__((ext_vector_type(8)))  __bf16 bf16x8;
typedef __attribute__((ext_vector_type(4)))  __bf16 bf16x4;
typedef __attribute__((ext_vector_type(8)))  float  f32x8;

#define H_DIM 2048
#define I_DIM 1024
#define KQ    8
#define E_DIM 128
#define EPS   1e-8f

// ---------------- WMMA wrapper ----------------
static __device__ __forceinline__ f32x8 wmma_bf16(bf16x16 a, bf16x16 b, f32x8 c) {
  return __builtin_amdgcn_wmma_f32_16x16x32_bf16(false, a, false, b, (short)0, c,
                                                 false, false);
}

// ---------------- fragment loaders (16-bit, contiguous k = 16*half + j per lane) ----------------
// A (16x32, row-major source): lane<16 holds row r, K = k0..k0+15;
// lane>=16 holds row r, K = k0+16..k0+31.
static __device__ __forceinline__ bf16x16
load_a_frag(const bf16_t* __restrict__ A, int lda, int row0, int k0) {
  const int lane = threadIdx.x & 31;
  const int half = lane >> 4;
  const int r    = lane & 15;
  const bf16_t* p = A + (size_t)(row0 + r) * lda + (k0 + half * 16);
  return *(const bf16x16*)(p);
}

// B (32x16) from K-major (pre-transposed) weights BT[N][Kdim]:
// lane holds column (lane&15), K = k0 + (lane>>4)*16 + j  (16 contiguous).
static __device__ __forceinline__ bf16x16
load_b_frag(const bf16_t* __restrict__ BT, int ldb, int col0, int k0) {
  const int lane = threadIdx.x & 31;
  const int n  = col0 + (lane & 15);
  const int kb = k0 + (lane >> 4) * 16;
  return *(const bf16x16*)(BT + (size_t)n * ldb + kb);
}

// ---------------- 2x2 register blocking helpers ----------------
struct Frag2 { bf16x16 f0, f1; };
struct Acc22 { f32x8 c00, c01, c10, c11; };

static __device__ __forceinline__ Frag2
load_a2(const bf16_t* __restrict__ A, int lda, int m0, int k) {
  Frag2 r; r.f0 = load_a_frag(A, lda, m0, k); r.f1 = load_a_frag(A, lda, m0 + 16, k);
  return r;
}
static __device__ __forceinline__ Frag2
load_b2(const bf16_t* __restrict__ BT, int ldb, int n0, int k) {
  Frag2 r; r.f0 = load_b_frag(BT, ldb, n0, k); r.f1 = load_b_frag(BT, ldb, n0 + 16, k);
  return r;
}
static __device__ __forceinline__ void mma22(Acc22& a, const Frag2& fa, const Frag2& fb) {
  a.c00 = wmma_bf16(fa.f0, fb.f0, a.c00);
  a.c01 = wmma_bf16(fa.f0, fb.f1, a.c01);
  a.c10 = wmma_bf16(fa.f1, fb.f0, a.c10);
  a.c11 = wmma_bf16(fa.f1, fb.f1, a.c11);
}

// C/D (16x16 f32): lane col = col0+(lane&15), rows = row0 + (lane>>4)*8 + v.
static __device__ __forceinline__ void
store_c_f32(float* __restrict__ C, int ldc, int row0, int col0, f32x8 c) {
  const int lane = threadIdx.x & 31;
  const int col  = col0 + (lane & 15);
  const int r0   = row0 + (lane >> 4) * 8;
#pragma unroll
  for (int v = 0; v < 8; ++v) C[(size_t)(r0 + v) * ldc + col] = c[v];
}

static __device__ __forceinline__ void
store_c_bf16(bf16_t* __restrict__ C, int ldc, int row0, int col0, f32x8 c) {
  const int lane = threadIdx.x & 31;
  const int col  = col0 + (lane & 15);
  const int r0   = row0 + (lane >> 4) * 8;
#pragma unroll
  for (int v = 0; v < 8; ++v) C[(size_t)(r0 + v) * ldc + col] = (bf16_t)c[v];
}

static __device__ __forceinline__ float silu_f(float g) {
  return g / (1.0f + __expf(-g));
}

static __device__ __forceinline__ void
store_silu_tile(bf16_t* __restrict__ out, int ldc, int row0, int col0,
                f32x8 g, f32x8 u) {
  const int lane = threadIdx.x & 31;
  const int col  = col0 + (lane & 15);
  const int r0   = row0 + (lane >> 4) * 8;
#pragma unroll
  for (int v = 0; v < 8; ++v)
    out[(size_t)(r0 + v) * ldc + col] = (bf16_t)(silu_f(g[v]) * u[v]);
}

// wave tile inside a 64x128 block tile: 8 waves = 2(M) x 4(N), 32x32 each.
static __device__ __forceinline__ void wave_tile(int& tm, int& tn) {
  const int wave = threadIdx.x >> 5;
  tm = (wave & 1) * 32;
  tn = (wave >> 1) * 32;
}

// ---------------- conversion / transpose kernels ----------------
__global__ void k_cvt_bf16x4(const float* __restrict__ in,
                             bf16_t* __restrict__ out, size_t n4) {
  size_t i = (size_t)blockIdx.x * blockDim.x + threadIdx.x;
  if (i >= n4) return;
  float4 f = ((const float4*)in)[i];
  bf16x4 b; b[0] = (bf16_t)f.x; b[1] = (bf16_t)f.y; b[2] = (bf16_t)f.z; b[3] = (bf16_t)f.w;
  ((bf16x4*)out)[i] = b;
}

// in: [R, C] f32 row-major  ->  out: [C, R] bf16 (K-major weights)
__global__ void k_transpose_bf16(const float* __restrict__ in,
                                 bf16_t* __restrict__ out, int R, int C) {
  __shared__ float tile[32][33];
  const int bx = blockIdx.x * 32;   // over C
  const int by = blockIdx.y * 32;   // over R
  const int tx = threadIdx.x, ty = threadIdx.y;
#pragma unroll
  for (int i = 0; i < 32; i += 8)
    tile[ty + i][tx] = in[(size_t)(by + ty + i) * C + (bx + tx)];
  __syncthreads();
#pragma unroll
  for (int i = 0; i < 32; i += 8)
    out[(size_t)(bx + ty + i) * R + (by + tx)] = (bf16_t)tile[tx][ty + i];
}

// ---------------- generic GEMM (router logits) ----------------
__global__ void __launch_bounds__(256)
k_gemm_bf16(const bf16_t* __restrict__ A, const bf16_t* __restrict__ BT,
            float* __restrict__ C, int Kdim, int ldc) {
  int tm, tn; wave_tile(tm, tn);
  const int m0 = blockIdx.x * 64 + tm;
  const int n0 = blockIdx.y * 128 + tn;
  Acc22 acc = {};
  Frag2 fa = load_a2(A, Kdim, m0, 0);
  Frag2 fb = load_b2(BT, Kdim, n0, 0);
  for (int k = 0; k < Kdim - 32; k += 32) {
    Frag2 nfa = load_a2(A, Kdim, m0, k + 32);
    Frag2 nfb = load_b2(BT, Kdim, n0, k + 32);
    mma22(acc, fa, fb);
    fa = nfa; fb = nfb;
  }
  mma22(acc, fa, fb);
  store_c_f32(C, ldc, m0,      n0,      acc.c00);
  store_c_f32(C, ldc, m0,      n0 + 16, acc.c01);
  store_c_f32(C, ldc, m0 + 16, n0,      acc.c10);
  store_c_f32(C, ldc, m0 + 16, n0 + 16, acc.c11);
}

// ---------------- router post: softmax + top4 + fold mix into cmix[T,8] ----
__global__ void k_router_post(const float* __restrict__ logits,
                              const float* __restrict__ amplitudes, // [E,K,2]
                              const float* __restrict__ expert_scale,
                              float* __restrict__ cmix, int T) {
  const int wave = (int)((blockIdx.x * (size_t)blockDim.x + threadIdx.x) >> 5);
  const int lane = threadIdx.x & 31;
  if (wave >= T) return;
  const float* lg = logits + (size_t)wave * E_DIM;

  float l[4];
#pragma unroll
  for (int j = 0; j < 4; ++j) l[j] = lg[lane + 32 * j];

  float mx = fmaxf(fmaxf(l[0], l[1]), fmaxf(l[2], l[3]));
#pragma unroll
  for (int off = 16; off; off >>= 1) mx = fmaxf(mx, __shfl_xor(mx, off, 32));

  float e[4], s = 0.f;
#pragma unroll
  for (int j = 0; j < 4; ++j) { e[j] = __expf(l[j] - mx); s += e[j]; }
#pragma unroll
  for (int off = 16; off; off >>= 1) s += __shfl_xor(s, off, 32);

  float p[4];
#pragma unroll
  for (int j = 0; j < 4; ++j) p[j] = e[j] / s;

  int   selIdx[4];
  float selP[4];
#pragma unroll
  for (int t = 0; t < 4; ++t) {
    float best = -1.f; int bi = E_DIM;
#pragma unroll
    for (int j = 0; j < 4; ++j)
      if (p[j] > best) { best = p[j]; bi = lane + 32 * j; }
#pragma unroll
    for (int off = 16; off; off >>= 1) {
      float ob = __shfl_xor(best, off, 32);
      int   oi = __shfl_xor(bi, off, 32);
      if (ob > best || (ob == best && oi < bi)) { best = ob; bi = oi; }
    }
    selIdx[t] = bi; selP[t] = best;
#pragma unroll
    for (int j = 0; j < 4; ++j)
      if (lane + 32 * j == bi) p[j] = -1.f;
  }
  const float ssum = selP[0] + selP[1] + selP[2] + selP[3];

  if (lane < KQ) {
    const int k = lane;
    float acc = 0.f;
#pragma unroll
    for (int t = 0; t < 4; ++t) {
      const int ei = selIdx[t];
      const float coeff = selP[t] / (ssum + EPS) * expert_scale[ei];
      const float* amp = amplitudes + (size_t)ei * (KQ * 2);
      float denom = 0.f, pk = 0.f;
#pragma unroll
      for (int kk = 0; kk < KQ; ++kk) {
        const float a0 = amp[kk * 2], a1 = amp[kk * 2 + 1];
        const float q = a0 * a0 + a1 * a1;
        denom += q;
        if (kk == k) pk = q;
      }
      acc += coeff * pk / (denom + EPS);
    }
    cmix[(size_t)wave * KQ + k] = acc;
  }
}

// ---------------- shared experts hidden: silu(x@Wg)*(x@Wu) -> bf16 [T,2I] --
__global__ void __launch_bounds__(256)
k_shared_hidden(const bf16_t* __restrict__ X, const bf16_t* __restrict__ GT,
                const bf16_t* __restrict__ UT, bf16_t* __restrict__ Hout,
                int N /*=2I*/) {
  int tm, tn; wave_tile(tm, tn);
  const int m0 = blockIdx.x * 64 + tm;
  const int n0 = blockIdx.y * 128 + tn;
  Acc22 g = {}, u = {};
  Frag2 fa = load_a2(X, H_DIM, m0, 0);
  Frag2 bg = load_b2(GT, H_DIM, n0, 0);
  Frag2 bu = load_b2(UT, H_DIM, n0, 0);
  for (int k = 0; k < H_DIM - 32; k += 32) {
    Frag2 nfa = load_a2(X, H_DIM, m0, k + 32);
    Frag2 nbg = load_b2(GT, H_DIM, n0, k + 32);
    Frag2 nbu = load_b2(UT, H_DIM, n0, k + 32);
    mma22(g, fa, bg);
    mma22(u, fa, bu);
    fa = nfa; bg = nbg; bu = nbu;
  }
  mma22(g, fa, bg);
  mma22(u, fa, bu);
  store_silu_tile(Hout, N, m0,      n0,      g.c00, u.c00);
  store_silu_tile(Hout, N, m0,      n0 + 16, g.c01, u.c01);
  store_silu_tile(Hout, N, m0 + 16, n0,      g.c10, u.c10);
  store_silu_tile(Hout, N, m0 + 16, n0 + 16, g.c11, u.c11);
}

// ---------------- routed basis + cmix-fold: comb[T,I] (bf16) ---------------
static __device__ __forceinline__ void
fold_comb(f32x8& c, f32x8 g, f32x8 u, const float* __restrict__ cmix,
          int row0, int kq) {
  const int lane = threadIdx.x & 31;
  const int r0   = row0 + (lane >> 4) * 8;
#pragma unroll
  for (int v = 0; v < 8; ++v) {
    const float w = cmix[(size_t)(r0 + v) * KQ + kq];
    c[v] += w * silu_f(g[v]) * u[v];
  }
}

__global__ void __launch_bounds__(256)
k_basis_combine(const bf16_t* __restrict__ X, const bf16_t* __restrict__ GT,
                const bf16_t* __restrict__ UT, const float* __restrict__ cmix,
                bf16_t* __restrict__ comb) {
  int tm, tn; wave_tile(tm, tn);
  const int m0 = blockIdx.x * 64 + tm;
  const int i0 = blockIdx.y * 128 + tn;          // column within I
  Acc22 c = {};
  for (int kq = 0; kq < KQ; ++kq) {
    const int n0 = kq * I_DIM + i0;              // column within K*I
    Acc22 g = {}, u = {};
    Frag2 fa = load_a2(X, H_DIM, m0, 0);
    Frag2 bg = load_b2(GT, H_DIM, n0, 0);
    Frag2 bu = load_b2(UT, H_DIM, n0, 0);
    for (int k = 0; k < H_DIM - 32; k += 32) {
      Frag2 nfa = load_a2(X, H_DIM, m0, k + 32);
      Frag2 nbg = load_b2(GT, H_DIM, n0, k + 32);
      Frag2 nbu = load_b2(UT, H_DIM, n0, k + 32);
      mma22(g, fa, bg);
      mma22(u, fa, bu);
      fa = nfa; bg = nbg; bu = nbu;
    }
    mma22(g, fa, bg);
    mma22(u, fa, bu);
    fold_comb(c.c00, g.c00, u.c00, cmix, m0,      kq);
    fold_comb(c.c01, g.c01, u.c01, cmix, m0,      kq);
    fold_comb(c.c10, g.c10, u.c10, cmix, m0 + 16, kq);
    fold_comb(c.c11, g.c11, u.c11, cmix, m0 + 16, kq);
  }
  store_c_bf16(comb, I_DIM, m0,      i0,      c.c00);
  store_c_bf16(comb, I_DIM, m0,      i0 + 16, c.c01);
  store_c_bf16(comb, I_DIM, m0 + 16, i0,      c.c10);
  store_c_bf16(comb, I_DIM, m0 + 16, i0 + 16, c.c11);
}

// ---------------- fused down-projection: out = comb@Wd + shh@SWd ----------
__global__ void __launch_bounds__(256)
k_down_out(const bf16_t* __restrict__ comb, const bf16_t* __restrict__ WDT,
           const bf16_t* __restrict__ shh,  const bf16_t* __restrict__ SWDT,
           float* __restrict__ out) {
  int tm, tn; wave_tile(tm, tn);
  const int m0 = blockIdx.x * 64 + tm;
  const int n0 = blockIdx.y * 128 + tn;          // over H
  Acc22 acc = {};
  {
    Frag2 fa = load_a2(comb, I_DIM, m0, 0);
    Frag2 fb = load_b2(WDT, I_DIM, n0, 0);
    for (int k = 0; k < I_DIM - 32; k += 32) {   // routed part (K = I)
      Frag2 nfa = load_a2(comb, I_DIM, m0, k + 32);
      Frag2 nfb = load_b2(WDT, I_DIM, n0, k + 32);
      mma22(acc, fa, fb);
      fa = nfa; fb = nfb;
    }
    mma22(acc, fa, fb);
  }
  {
    Frag2 fa = load_a2(shh, 2 * I_DIM, m0, 0);
    Frag2 fb = load_b2(SWDT, 2 * I_DIM, n0, 0);
    for (int k = 0; k < 2 * I_DIM - 32; k += 32) {  // shared part (K = 2I)
      Frag2 nfa = load_a2(shh, 2 * I_DIM, m0, k + 32);
      Frag2 nfb = load_b2(SWDT, 2 * I_DIM, n0, k + 32);
      mma22(acc, fa, fb);
      fa = nfa; fb = nfb;
    }
    mma22(acc, fa, fb);
  }
  store_c_f32(out, H_DIM, m0,      n0,      acc.c00);
  store_c_f32(out, H_DIM, m0,      n0 + 16, acc.c01);
  store_c_f32(out, H_DIM, m0 + 16, n0,      acc.c10);
  store_c_f32(out, H_DIM, m0 + 16, n0 + 16, acc.c11);
}

// ---------------------------------------------------------------------------
extern "C" void kernel_launch(void* const* d_in, const int* in_sizes, int n_in,
                              void* d_out, int out_size, void* d_ws, size_t ws_size,
                              hipStream_t stream) {
  (void)n_in; (void)out_size; (void)ws_size;
  const float* x            = (const float*)d_in[0];
  const float* w_router     = (const float*)d_in[1];
  const float* w_gate       = (const float*)d_in[2];
  const float* w_up         = (const float*)d_in[3];
  const float* w_down       = (const float*)d_in[4];
  const float* amplitudes   = (const float*)d_in[5];
  const float* expert_scale = (const float*)d_in[6];
  const float* swg          = (const float*)d_in[7];
  const float* swu          = (const float*)d_in[8];
  const float* swd          = (const float*)d_in[9];
  float* out = (float*)d_out;

  const int T = in_sizes[0] / H_DIM;   // 8192 for the reference shapes

  // ---- workspace layout (bf16 operand copies + intermediates) ----
  char* ws = (char*)d_ws;
  size_t off = 0;
  auto alloc = [&](size_t bytes) -> char* {
    char* p = ws + off;
    off = (off + bytes + 255) & ~(size_t)255;
    return p;
  };
  bf16_t* xb   = (bf16_t*)alloc((size_t)T * H_DIM * 2);
  bf16_t* wgT  = (bf16_t*)alloc((size_t)KQ * I_DIM * H_DIM * 2);
  bf16_t* wuT  = (bf16_t*)alloc((size_t)KQ * I_DIM * H_DIM * 2);
  bf16_t* wdT  = (bf16_t*)alloc((size_t)H_DIM * I_DIM * 2);
  bf16_t* swgT = (bf16_t*)alloc((size_t)2 * I_DIM * H_DIM * 2);
  bf16_t* swuT = (bf16_t*)alloc((size_t)2 * I_DIM * H_DIM * 2);
  bf16_t* swdT = (bf16_t*)alloc((size_t)H_DIM * 2 * I_DIM * 2);
  bf16_t* wrT  = (bf16_t*)alloc((size_t)E_DIM * H_DIM * 2);
  float*  logits = (float*)alloc((size_t)T * E_DIM * 4);
  float*  cmix   = (float*)alloc((size_t)T * KQ * 4);
  bf16_t* comb   = (bf16_t*)alloc((size_t)T * I_DIM * 2);
  bf16_t* shh    = (bf16_t*)alloc((size_t)T * 2 * I_DIM * 2);

  // ---- stage 0: fp32 -> bf16 conversions (x) and K-major weight transposes
  {
    const size_t n4 = (size_t)T * H_DIM / 4;
    const int blk = 256;
    k_cvt_bf16x4<<<(unsigned)((n4 + blk - 1) / blk), blk, 0, stream>>>(x, xb, n4);
  }
  dim3 tb(32, 8);
  k_transpose_bf16<<<dim3((KQ * I_DIM) / 32, H_DIM / 32), tb, 0, stream>>>(w_gate, wgT, H_DIM, KQ * I_DIM);
  k_transpose_bf16<<<dim3((KQ * I_DIM) / 32, H_DIM / 32), tb, 0, stream>>>(w_up,   wuT, H_DIM, KQ * I_DIM);
  k_transpose_bf16<<<dim3(H_DIM / 32, I_DIM / 32),        tb, 0, stream>>>(w_down, wdT, I_DIM, H_DIM);
  k_transpose_bf16<<<dim3((2 * I_DIM) / 32, H_DIM / 32),  tb, 0, stream>>>(swg, swgT, H_DIM, 2 * I_DIM);
  k_transpose_bf16<<<dim3((2 * I_DIM) / 32, H_DIM / 32),  tb, 0, stream>>>(swu, swuT, H_DIM, 2 * I_DIM);
  k_transpose_bf16<<<dim3(H_DIM / 32, (2 * I_DIM) / 32),  tb, 0, stream>>>(swd, swdT, 2 * I_DIM, H_DIM);
  k_transpose_bf16<<<dim3(E_DIM / 32, H_DIM / 32),        tb, 0, stream>>>(w_router, wrT, H_DIM, E_DIM);

  // ---- stage 1: router logits (WMMA GEMM) + softmax/top4/cmix fold ----
  k_gemm_bf16<<<dim3(T / 64, E_DIM / 128), 256, 0, stream>>>(xb, wrT, logits, H_DIM, E_DIM);
  k_router_post<<<(T * 32 + 255) / 256, 256, 0, stream>>>(logits, amplitudes, expert_scale, cmix, T);

  // ---- stage 2: shared-expert hidden (silu-gated, WMMA) ----
  k_shared_hidden<<<dim3(T / 64, (2 * I_DIM) / 128), 256, 0, stream>>>(xb, swgT, swuT, shh, 2 * I_DIM);

  // ---- stage 3: routed basis GEMMs folded with cmix (WMMA) ----
  k_basis_combine<<<dim3(T / 64, I_DIM / 128), 256, 0, stream>>>(xb, wgT, wuT, cmix, comb);

  // ---- stage 4: fused down-projection (routed + shared, WMMA) ----
  k_down_out<<<dim3(T / 64, H_DIM / 128), 256, 0, stream>>>(comb, wdT, shh, swdT, out);
}
